// CausalSelfAttention_31731218382898
// MI455X (gfx1250) — matrix operand order repaired and checked by reading, not compiled, not run on hardware
//
#include <hip/hip_runtime.h>
#include <stdint.h>

#define BB 4
#define NN 2048
#define DD 1024
#define HH 16
#define HD 64
#define E3 192

typedef __attribute__((ext_vector_type(16))) __bf16 v16bf;
typedef __attribute__((ext_vector_type(8)))  float  v8f;
typedef __attribute__((ext_vector_type(4)))  int    v4i;

typedef __attribute__((address_space(1))) v4i* gptr_v4i;
typedef __attribute__((address_space(3))) v4i* lptr_v4i;

union ABReg {
    uint4 u4[2];
    v16bf v;
};

#if defined(__has_builtin)
#  if __has_builtin(__builtin_amdgcn_global_load_async_to_lds_b128)
#    define HAVE_ASYNC_LDS 1
#  endif
#endif
#ifndef HAVE_ASYNC_LDS
#  define HAVE_ASYNC_LDS 0
#endif

#if HAVE_ASYNC_LDS
#  if __has_builtin(__builtin_amdgcn_s_wait_asynccnt)
#    define WAIT_ASYNC(N) __builtin_amdgcn_s_wait_asynccnt(N)
#  else
#    define WAIT_ASYNC(N) asm volatile("s_wait_asynccnt " #N ::: "memory")
#  endif
#else
#  define WAIT_ASYNC(N) do {} while (0)
#endif

// 16-byte global -> LDS copy (async on gfx1250)
__device__ __forceinline__ void cp16(const unsigned short* g, unsigned short* l) {
#if HAVE_ASYNC_LDS
    __builtin_amdgcn_global_load_async_to_lds_b128(
        (gptr_v4i)(uintptr_t)g,
        (lptr_v4i)(unsigned int)(uintptr_t)l,
        0, 0);
#else
    *(uint4*)l = *(const uint4*)g;
#endif
}

__device__ __forceinline__ unsigned short f32_bf16(float f) {
    unsigned int u = __float_as_uint(f);
    u = (u + 0x7FFFu + ((u >> 16) & 1u)) >> 16;
    return (unsigned short)u;
}

// ---------------------------------------------------------------------------
// Kernel 1: x (fp32) -> bf16
// ---------------------------------------------------------------------------
__global__ void cvt_x_bf16(const float* __restrict__ x,
                           unsigned short* __restrict__ xb, int n) {
    int i = blockIdx.x * blockDim.x + threadIdx.x;
    if (i < n) xb[i] = f32_bf16(x[i]);
}

// ---------------------------------------------------------------------------
// Kernel 2: W[h][d][e] (fp32) -> Wt[h][e][d] (bf16)
// ---------------------------------------------------------------------------
__global__ void transpose_W(const float* __restrict__ W,
                            unsigned short* __restrict__ Wt, int n) {
    int i = blockIdx.x * blockDim.x + threadIdx.x;
    if (i >= n) return;
    int d = i & (DD - 1);
    int e = (i >> 10) % E3;
    int h = i / (E3 * DD);
    Wt[i] = f32_bf16(W[((size_t)h * DD + d) * E3 + e]);
}

// ---------------------------------------------------------------------------
// Kernel 3: fused per-head KQV projection. Async global->LDS double
// buffering with MANUAL unroll-by-2 so bufsel is a literal constant in
// every issue/body call: both LDS buffers are compile-time offsets, no
// runtime buffer select, one canonical register assignment per instance.
// ---------------------------------------------------------------------------
__global__ void __launch_bounds__(256)
proj_kqv(const unsigned short* __restrict__ xb,
         const unsigned short* __restrict__ Wt,
         const float* __restrict__ bias,
         unsigned short* __restrict__ Qb,
         unsigned short* __restrict__ Kb,
         unsigned short* __restrict__ Vt) {
    alignas(16) __shared__ unsigned short Xs[2][128][40];  // 20 KB
    alignas(16) __shared__ unsigned short Ws[2][192][40];  // 30 KB

    const int tid     = threadIdx.x;
    const int lane    = tid & 31;
    const int wave    = tid >> 5;
    const int bh      = blockIdx.y;
    const int b       = bh >> 4;
    const int h       = bh & 15;
    const int brow    = blockIdx.x * 128;
    const int rowgrp  = wave & 3;        // 32-row group within block
    const int colgrp  = wave >> 2;       // 0/1 -> e-tiles [0..5] / [6..11]
    const int halfsel = lane >> 4;
    const int col     = lane & 15;
    const int kb0     = halfsel ? 8 : 0;

    // staging addresses hoisted out of the K loop
    const int srow = tid >> 2;           // 0..63
    const int scc8 = (tid & 3) * 8;
    const unsigned short* xg0 =
        xb + ((size_t)b * NN + brow + srow) * DD + scc8;
    const unsigned short* wg0 =
        Wt + ((size_t)h * E3 + srow) * DD + scc8;

    auto issue = [&](int kc, int bufsel) {
        cp16(xg0 + kc,             &Xs[bufsel][srow][scc8]);
        cp16(xg0 + 64 * DD + kc,   &Xs[bufsel][srow + 64][scc8]);
        cp16(wg0 + kc,             &Ws[bufsel][srow][scc8]);
        cp16(wg0 + 64 * DD + kc,   &Ws[bufsel][srow + 64][scc8]);
        cp16(wg0 + 128 * DD + kc,  &Ws[bufsel][srow + 128][scc8]);
    };

    v8f acc[2][6];
#pragma unroll
    for (int rt = 0; rt < 2; ++rt)
#pragma unroll
        for (int t = 0; t < 6; ++t) acc[rt][t] = {};

    auto body = [&](int bufsel) {   // always called with a literal constant
        ABReg a0, a1, bm[6];
        const unsigned short* xr0 = &Xs[bufsel][rowgrp * 32 + col][kb0];
        const unsigned short* xr1 = xr0 + 16 * 40;
        a0.u4[0] = *(const uint4*)xr0;
        a0.u4[1] = *(const uint4*)(xr0 + 16);
        a1.u4[0] = *(const uint4*)xr1;
        a1.u4[1] = *(const uint4*)(xr1 + 16);
#pragma unroll
        for (int t = 0; t < 6; ++t) {
            const unsigned short* wr =
                &Ws[bufsel][colgrp * 96 + t * 16 + col][kb0];
            bm[t].u4[0] = ((const uint4*)wr)[0];
            bm[t].u4[1] = ((const uint4*)(wr + 16))[0];
        }
#pragma unroll
        for (int t = 0; t < 6; ++t) {
            acc[0][t] = __builtin_amdgcn_wmma_f32_16x16x32_bf16(
                false, a0.v, false, bm[t].v, (short)0, acc[0][t], false, false);
            acc[1][t] = __builtin_amdgcn_wmma_f32_16x16x32_bf16(
                false, a1.v, false, bm[t].v, (short)0, acc[1][t], false, false);
        }
    };

    // NK = 32 stages. Manual unroll-by-2, literal bufsel everywhere.
    // Loop computes stages 0..29 (15 iterations x 2); tail: 30, 31.
    issue(0, 0);
    int kc = 32;                               // next stage byte... element offset
    for (int it = 0; it < 15; ++it) {
        issue(kc, 1);  kc += 32;
        WAIT_ASYNC(5);
        __syncthreads();
        body(0);
        __syncthreads();
        issue(kc, 0);  kc += 32;
        WAIT_ASYNC(5);
        __syncthreads();
        body(1);
        __syncthreads();
    }
    issue(kc, 1);                              // stage 31
    WAIT_ASYNC(5);
    __syncthreads();
    body(0);                                   // stage 30
    __syncthreads();
    WAIT_ASYNC(0);
    __syncthreads();
    body(1);                                   // stage 31

#pragma unroll
    for (int t = 0; t < 6; ++t) {
        const int tg = colgrp * 6 + t;       // global e-tile index 0..11
        const int e  = tg * 16 + col;
        const float bv = bias[h * E3 + e];
#pragma unroll
        for (int rt = 0; rt < 2; ++rt) {
#pragma unroll
            for (int r = 0; r < 8; ++r) {
                const int row = brow + rowgrp * 32 + rt * 16 + r + 8 * halfsel;
                const unsigned short us = f32_bf16(acc[rt][t][r] + bv);
                if (tg < 4) {
                    Kb[((size_t)bh * NN + row) * HD + e] = us;
                } else if (tg < 8) {
                    Qb[((size_t)bh * NN + row) * HD + (e - 64)] = us;
                } else {
                    Vt[((size_t)bh * HD + (e - 128)) * NN + row] = us;
                }
            }
        }
    }
}

// ---------------------------------------------------------------------------
// Kernel 4: causal flash attention, one wave = TWO adjacent 16-row q tiles
// (32 rows). K/V loads for each 32-key block are shared by both q tiles:
// 16 WMMAs per block for the same 8x32B of loads. K tiles ping-pong between
// two register buffers via a 2-unrolled loop (no guarded copies).
// ---------------------------------------------------------------------------
__global__ void __launch_bounds__(128)
flash_attn(const unsigned short* __restrict__ Qb,
           const unsigned short* __restrict__ Kb,
           const unsigned short* __restrict__ Vt,
           float* __restrict__ out) {
    __shared__ unsigned short pbuf[4][2][16][32];  // 8 KB

    const int lane  = threadIdx.x & 31;
    const int wave  = threadIdx.x >> 5;
    const int tile  = blockIdx.x * 4 + wave;       // 32-row pair tile
    const int bh    = tile >> 6;                   // tile / (N/32)
    const int qt    = tile & 63;
    const int qrow0 = qt * 32;
    const int b     = bh >> 4;
    const int h     = bh & 15;

    const int halfsel = lane >> 4;
    const int col     = lane & 15;
    const int kb0     = halfsel ? 8 : 0;

    const unsigned short* Qp = Qb + (size_t)bh * NN * HD;
    const unsigned short* Kp = Kb + (size_t)bh * NN * HD;
    const unsigned short* Vp = Vt + (size_t)bh * HD * NN;

    ABReg aq[2][2];
#pragma unroll
    for (int qi = 0; qi < 2; ++qi) {
        const unsigned short* qr = Qp + (size_t)(qrow0 + qi * 16 + col) * HD;
        aq[qi][0].u4[0] = *(const uint4*)(qr + 16 * halfsel);
        aq[qi][0].u4[1] = *(const uint4*)(qr + 16 * halfsel + 8);
        aq[qi][1].u4[0] = *(const uint4*)(qr + 32 + 16 * halfsel);
        aq[qi][1].u4[1] = *(const uint4*)(qr + 32 + 16 * halfsel + 8);
    }

    v8f o[2][4];
    float mrow[2][8], lrow[2][8];
#pragma unroll
    for (int qi = 0; qi < 2; ++qi) {
#pragma unroll
        for (int t = 0; t < 4; ++t) o[qi][t] = {};
#pragma unroll
        for (int r = 0; r < 8; ++r) { mrow[qi][r] = -1e30f; lrow[qi][r] = 0.0f; }
    }

    const float scale = 0.125f;
    const float LOG2E = 1.4426950408889634f;

    auto loadK = [&](ABReg (&dst)[4], int kb) {
        const unsigned short* kr0 = Kp + (size_t)(kb + col) * HD + 16 * halfsel;
        const unsigned short* kr1 = Kp + (size_t)(kb + 16 + col) * HD + 16 * halfsel;
        dst[0].u4[0] = ((const uint4*)kr0)[0];
        dst[0].u4[1] = ((const uint4*)kr0)[1];
        dst[1].u4[0] = ((const uint4*)(kr0 + 32))[0];
        dst[1].u4[1] = ((const uint4*)(kr0 + 32))[1];
        dst[2].u4[0] = ((const uint4*)kr1)[0];
        dst[2].u4[1] = ((const uint4*)kr1)[1];
        dst[3].u4[0] = ((const uint4*)(kr1 + 32))[0];
        dst[3].u4[1] = ((const uint4*)(kr1 + 32))[1];
    };

    const int nkb = qt + 1;   // keys 0 .. qrow0+31 in 32-key blocks

    auto step = [&](int kblk, ABReg (&cur)[4], ABReg (&nxt)[4]) {
        const int kb = kblk * 32;

        // V tiles for this block (shared by both q tiles), issued early
        ABReg vcur[4];
#pragma unroll
        for (int t = 0; t < 4; ++t) {
            const unsigned short* vp =
                Vp + (size_t)(t * 16 + col) * NN + kb + 16 * halfsel;
            vcur[t].u4[0] = ((const uint4*)vp)[0];
            vcur[t].u4[1] = ((const uint4*)vp)[1];
        }
        // prefetch next K block into the other buffer
        if (kblk + 1 < nkb) loadK(nxt, kb + 32);

#pragma unroll
        for (int qi = 0; qi < 2; ++qi) {
            const int qrow = qrow0 + qi * 16;

            v8f s0 = {}, s1 = {};
            s0 = __builtin_amdgcn_wmma_f32_16x16x32_bf16(false, aq[qi][0].v, false, cur[0].v, (short)0, s0, false, false);
            s0 = __builtin_amdgcn_wmma_f32_16x16x32_bf16(false, aq[qi][1].v, false, cur[1].v, (short)0, s0, false, false);
            s1 = __builtin_amdgcn_wmma_f32_16x16x32_bf16(false, aq[qi][0].v, false, cur[2].v, (short)0, s1, false, false);
            s1 = __builtin_amdgcn_wmma_f32_16x16x32_bf16(false, aq[qi][1].v, false, cur[3].v, (short)0, s1, false, false);

            const bool need_mask = (kb + 31) > qrow;
#pragma unroll
            for (int r = 0; r < 8; ++r) {
                const int qg = qrow + r + 8 * halfsel;
                float v0 = s0[r] * scale;
                float v1 = s1[r] * scale;
                if (need_mask) {
                    if (kb + col > qg)      v0 = -1e30f;
                    if (kb + 16 + col > qg) v1 = -1e30f;
                }
                s0[r] = v0;
                s1[r] = v1;
            }

#pragma unroll
            for (int r = 0; r < 8; ++r) {
                float bm = fmaxf(s0[r], s1[r]);
                bm = fmaxf(bm, __shfl_xor(bm, 1, 32));
                bm = fmaxf(bm, __shfl_xor(bm, 2, 32));
                bm = fmaxf(bm, __shfl_xor(bm, 4, 32));
                bm = fmaxf(bm, __shfl_xor(bm, 8, 32));
                const float mnew  = fmaxf(mrow[qi][r], bm);
                const float alpha = exp2f((mrow[qi][r] - mnew) * LOG2E);
                const float p0 = exp2f((s0[r] - mnew) * LOG2E);
                const float p1 = exp2f((s1[r] - mnew) * LOG2E);
                float rs = p0 + p1;
                rs += __shfl_xor(rs, 1, 32);
                rs += __shfl_xor(rs, 2, 32);
                rs += __shfl_xor(rs, 4, 32);
                rs += __shfl_xor(rs, 8, 32);
                lrow[qi][r] = lrow[qi][r] * alpha + rs;
                mrow[qi][r] = mnew;
                o[qi][0][r] *= alpha; o[qi][1][r] *= alpha;
                o[qi][2][r] *= alpha; o[qi][3][r] *= alpha;
                pbuf[wave][qi][r + 8 * halfsel][col]      = f32_bf16(p0);
                pbuf[wave][qi][r + 8 * halfsel][col + 16] = f32_bf16(p1);
            }
            asm volatile("s_wait_dscnt 0" ::: "memory");

            ABReg ap;
            const unsigned short* pp = &pbuf[wave][qi][col][0];
            ap.u4[0] = *(const uint4*)(pp + 16 * halfsel);
            ap.u4[1] = *(const uint4*)(pp + 16 * halfsel + 8);

            o[qi][0] = __builtin_amdgcn_wmma_f32_16x16x32_bf16(false, ap.v, false, vcur[0].v, (short)0, o[qi][0], false, false);
            o[qi][1] = __builtin_amdgcn_wmma_f32_16x16x32_bf16(false, ap.v, false, vcur[1].v, (short)0, o[qi][1], false, false);
            o[qi][2] = __builtin_amdgcn_wmma_f32_16x16x32_bf16(false, ap.v, false, vcur[2].v, (short)0, o[qi][2], false, false);
            o[qi][3] = __builtin_amdgcn_wmma_f32_16x16x32_bf16(false, ap.v, false, vcur[3].v, (short)0, o[qi][3], false, false);
        }
    };

    ABReg kA[4], kB[4];
    loadK(kA, 0);
    for (int kblk = 0; kblk < nkb; kblk += 2) {
        step(kblk, kA, kB);
        if (kblk + 1 < nkb) step(kblk + 1, kB, kA);
    }

    float* obase = out + (size_t)b * NN * (HH * HD) + (size_t)h * HD;
#pragma unroll
    for (int qi = 0; qi < 2; ++qi) {
#pragma unroll
        for (int r = 0; r < 8; ++r) {
            const int row = qrow0 + qi * 16 + r + 8 * halfsel;
            const float inv = 1.0f / lrow[qi][r];
            float* op = obase + (size_t)row * (HH * HD);
            op[0 * 16 + col] = o[qi][0][r] * inv;
            op[1 * 16 + col] = o[qi][1][r] * inv;
            op[2 * 16 + col] = o[qi][2][r] * inv;
            op[3 * 16 + col] = o[qi][3][r] * inv;
        }
    }
}

// ---------------------------------------------------------------------------
extern "C" void kernel_launch(void* const* d_in, const int* in_sizes, int n_in,
                              void* d_out, int out_size, void* d_ws, size_t ws_size,
                              hipStream_t stream) {
    const float* x    = (const float*)d_in[0];   // [B,N,D]
    const float* W    = (const float*)d_in[1];   // [H,D,192]
    const float* bias = (const float*)d_in[2];   // [H,192]
    float* out = (float*)d_out;                  // [B,N,H*64]

    char* ws = (char*)d_ws;
    size_t off = 0;
    unsigned short* xb = (unsigned short*)(ws + off); off += (size_t)BB * NN * DD * 2;
    unsigned short* Wt = (unsigned short*)(ws + off); off += (size_t)HH * E3 * DD * 2;
    unsigned short* Qb = (unsigned short*)(ws + off); off += (size_t)BB * HH * NN * HD * 2;
    unsigned short* Kb = (unsigned short*)(ws + off); off += (size_t)BB * HH * NN * HD * 2;
    unsigned short* Vt = (unsigned short*)(ws + off); off += (size_t)BB * HH * NN * HD * 2;
    (void)ws_size; (void)in_sizes; (void)n_in; (void)out_size;

    const int nx = BB * NN * DD;
    cvt_x_bf16<<<nx / 256, 256, 0, stream>>>(x, xb, nx);

    const int nw = HH * E3 * DD;
    transpose_W<<<nw / 256, 256, 0, stream>>>(W, Wt, nw);

    proj_kqv<<<dim3(NN / 128, BB * HH), 256, 0, stream>>>(xb, Wt, bias, Qb, Kb, Vt);

    const int ntiles = BB * HH * (NN / 32);      // 4096 pair-tiles
    flash_attn<<<ntiles / 4, 128, 0, stream>>>(Qb, Kb, Vt, out);
}
